// InteractionGNNBlock_83889301225977
// MI455X (gfx1250) — compile-verified, matrix operand order checked by reading, not run on hardware
//
#include <hip/hip_runtime.h>
#include <hip/hip_bf16.h>
#include <math.h>

// ---------------------------------------------------------------------------
// MI455X (gfx1250) fused GNN pipeline.
// All GEMMs run on v_wmma_f32_16x16x32_bf16, fp32 accumulate, bf16 operands.
// Double-buffered LDS staging with 128-bit global loads.
// ---------------------------------------------------------------------------

typedef __bf16 bf16;
typedef __attribute__((ext_vector_type(8)))  __bf16 v8bf;
typedef __attribute__((ext_vector_type(16))) __bf16 v16bf;
typedef __attribute__((ext_vector_type(8)))  float  v8f;

#define KCH 32   // K chunk per pipeline stage (one WMMA k-tile)

// Build a 16-element bf16 fragment from two contiguous 16-byte LDS loads.
// Layout per CDNA5 ISA 7.12.2: lane = row (A) / col (B), half = lane>>4
// selects K subsets {0..7,16..23} vs {8..15,24..31}.
__device__ inline v16bf ld_frag(const bf16* p) {
  v8bf lo = *(const v8bf*)p;
  v8bf hi = *(const v8bf*)(p + 16);
  return __builtin_shufflevector(lo, hi, 0,1,2,3,4,5,6,7,8,9,10,11,12,13,14,15);
}

// MODE 0: LayerNorm + ReLU (+optional residual)
// MODE 1: LayerNorm + tanh (+optional residual)
// MODE 2: bias only, then per-row L2 normalize over `ncols` valid columns
template<int TN, int WN, int MODE>
__global__ __launch_bounds__(256) void gemm_fused(
    const bf16* __restrict__ X,  const bf16* __restrict__ Wt,
    const float* __restrict__ bias, const float* __restrict__ gam,
    const float* __restrict__ bet,  const float* __restrict__ res,
    float* __restrict__ out_f, bf16* __restrict__ out_b,
    int M, int K, int ncols)
{
  constexpr int N    = WN * TN * 16;     // full output width owned by block
  constexpr int MB   = (8 / WN) * 16;    // rows per block (8 waves)
  constexpr int XOCT = MB * (KCH / 8);   // 16-byte octets per X chunk
  constexpr int WOCT = N  * (KCH / 8);   // 16-byte octets per W chunk
  constexpr int RX   = (XOCT + 255) / 256;
  constexpr int RW   = (WOCT + 255) / 256;

  __shared__ alignas(16) bf16 Xs[2][MB * KCH];
  __shared__ alignas(16) bf16 Ws[2][N  * KCH];
  __shared__ float redS[WN][MB];
  __shared__ float redQ[WN][MB];

  const int tid  = threadIdx.x;
  const int lane = tid & 31;
  const int w    = tid >> 5;
  const int wn   = w % WN;               // which N-slice this wave owns
  const int wm   = w / WN;               // which 16-row M tile
  const int rm   = lane & 15;
  const int half = lane >> 4;
  const int m0   = blockIdx.x * MB;

  uint4 rx[RX], rw[RW];

  auto ldX = [&](int kc) {
    #pragma unroll
    for (int j = 0; j < RX; ++j) {
      int o = tid + j * 256;
      uint4 v = {0u, 0u, 0u, 0u};
      if (o < XOCT) {
        int r  = o / (KCH / 8);
        int c8 = (o % (KCH / 8)) * 8;
        int gr = m0 + r;
        if (gr < M) {
          const bf16* p = X + (size_t)gr * K + kc + c8;
          v = *(const uint4*)p;
          if (j == 0 && kc + 2 * KCH < K) __builtin_prefetch(p + 2 * KCH, 0, 0);
        }
      }
      rx[j] = v;
    }
  };
  auto ldW = [&](int kc) {
    #pragma unroll
    for (int j = 0; j < RW; ++j) {
      int o = tid + j * 256;
      uint4 v = {0u, 0u, 0u, 0u};
      if (o < WOCT) {
        int n  = o / (KCH / 8);
        int c8 = (o % (KCH / 8)) * 8;
        v = *(const uint4*)(Wt + (size_t)n * K + kc + c8);
      }
      rw[j] = v;
    }
  };
  auto stX = [&](int cb) {
    #pragma unroll
    for (int j = 0; j < RX; ++j) {
      int o = tid + j * 256;
      if (o < XOCT) {
        int r  = o / (KCH / 8);
        int c8 = (o % (KCH / 8)) * 8;
        *(uint4*)(&Xs[cb][r * KCH + c8]) = rx[j];
      }
    }
  };
  auto stW = [&](int cb) {
    #pragma unroll
    for (int j = 0; j < RW; ++j) {
      int o = tid + j * 256;
      if (o < WOCT) {
        int n  = o / (KCH / 8);
        int c8 = (o % (KCH / 8)) * 8;
        *(uint4*)(&Ws[cb][n * KCH + c8]) = rw[j];
      }
    }
  };

  v8f zacc;
  #pragma unroll
  for (int e = 0; e < 8; ++e) zacc[e] = 0.f;
  v8f acc[TN];
  #pragma unroll
  for (int t = 0; t < TN; ++t) acc[t] = zacc;

  // prologue: fill buffer 0
  ldX(0); ldW(0);
  stX(0); stW(0);
  __syncthreads();

  // K is always a multiple of KCH (padded), so no tail handling.
  int cb = 0;
  for (int kc = 0; kc < K; kc += KCH, cb ^= 1) {
    const bool more = (kc + KCH) < K;
    if (more) { ldX(kc + KCH); ldW(kc + KCH); }   // global loads in flight

    v16bf a = ld_frag(&Xs[cb][(wm * 16 + rm) * KCH + half * 8]);
    #pragma unroll
    for (int t = 0; t < TN; ++t) {
      int n = (wn * TN + t) * 16 + rm;
      v16bf b = ld_frag(&Ws[cb][n * KCH + half * 8]);
      acc[t] = __builtin_amdgcn_wmma_f32_16x16x32_bf16(
          false, a, false, b, (short)0, acc[t], false, false);
    }

    if (more) { stX(cb ^ 1); stW(cb ^ 1); }       // fill alternate buffer
    __syncthreads();
  }

  // ---- epilogue -----------------------------------------------------------
  const int lrbase = wm * 16 + half * 8;   // first of the 8 rows this lane holds
  int ncol[TN];
  #pragma unroll
  for (int t = 0; t < TN; ++t) {
    ncol[t] = (wn * TN + t) * 16 + rm;
    float bv = (MODE == 2 && rm >= ncols) ? 0.f : bias[ncol[t]];
    #pragma unroll
    for (int i = 0; i < 8; ++i) acc[t][i] += bv;
  }

  if (MODE == 2) {
    // per-row L2 normalize over ncols valid columns (TN==1 path)
    float sq[8];
    #pragma unroll
    for (int i = 0; i < 8; ++i)
      sq[i] = (rm < ncols) ? acc[0][i] * acc[0][i] : 0.f;
    #pragma unroll
    for (int mk = 1; mk < 16; mk <<= 1)
      #pragma unroll
      for (int i = 0; i < 8; ++i) sq[i] += __shfl_xor(sq[i], mk, 32);
    #pragma unroll
    for (int i = 0; i < 8; ++i) {
      int rg = m0 + lrbase + i;
      float sc = 1.f / fmaxf(sqrtf(sq[i]), 1e-12f);
      float y  = acc[0][i] * sc;
      if (rm < ncols && rg < M) out_f[(size_t)rg * ncols + rm] = y;
    }
    return;
  }

  // LayerNorm statistics: partial sums per row, butterfly over the 16 lanes
  // sharing the same 8 rows (wave32), then cross-wave combine through LDS.
  float ps[8], pq[8];
  #pragma unroll
  for (int i = 0; i < 8; ++i) { ps[i] = 0.f; pq[i] = 0.f; }
  #pragma unroll
  for (int t = 0; t < TN; ++t)
    #pragma unroll
    for (int i = 0; i < 8; ++i) { ps[i] += acc[t][i]; pq[i] += acc[t][i] * acc[t][i]; }
  #pragma unroll
  for (int mk = 1; mk < 16; mk <<= 1)
    #pragma unroll
    for (int i = 0; i < 8; ++i) {
      ps[i] += __shfl_xor(ps[i], mk, 32);
      pq[i] += __shfl_xor(pq[i], mk, 32);
    }
  if (rm == 0) {
    #pragma unroll
    for (int i = 0; i < 8; ++i) {
      redS[wn][lrbase + i] = ps[i];
      redQ[wn][lrbase + i] = pq[i];
    }
  }
  __syncthreads();

  float mean[8], rstd[8];
  #pragma unroll
  for (int i = 0; i < 8; ++i) {
    float S = 0.f, Q = 0.f;
    #pragma unroll
    for (int q = 0; q < WN; ++q) { S += redS[q][lrbase + i]; Q += redQ[q][lrbase + i]; }
    float mu  = S / (float)N;
    float var = Q / (float)N - mu * mu;
    mean[i] = mu;
    rstd[i] = rsqrtf(var + 1e-5f);
  }

  #pragma unroll
  for (int t = 0; t < TN; ++t) {
    float gg = gam[ncol[t]], bb = bet[ncol[t]];
    #pragma unroll
    for (int i = 0; i < 8; ++i) {
      int rg = m0 + lrbase + i;
      float y = (acc[t][i] - mean[i]) * rstd[i] * gg + bb;
      y = (MODE == 1) ? tanhf(y) : fmaxf(y, 0.f);
      if (rg < M) {
        size_t o = (size_t)rg * N + ncol[t];
        if (res)   y += res[o];
        if (out_f) out_f[o] = y;
        if (out_b) out_b[o] = (bf16)y;
      }
    }
  }
}

// ---- helper kernels --------------------------------------------------------

// fp32 W[K,N] row-major -> bf16 Wt[Np][Kp] transposed, zero padded
__global__ void k_cvt_wt(const float* __restrict__ W, bf16* __restrict__ Wt,
                         int K, int N, int Kp, int Np) {
  int i = blockIdx.x * 256 + threadIdx.x;
  int total = Np * Kp;
  if (i >= total) return;
  int n = i / Kp, k = i % Kp;
  float v = (n < N && k < K) ? W[(size_t)k * N + n] : 0.f;
  Wt[i] = (bf16)v;
}

__global__ void k_node_in(const float* __restrict__ x, bf16* __restrict__ out, int Nn) {
  int i = blockIdx.x * 256 + threadIdx.x;
  if (i >= Nn * 32) return;
  int r = i >> 5, c = i & 31;
  out[i] = (bf16)((c < 3) ? x[r * 3 + c] : 0.f);
}

__global__ void k_edge_in(const float* __restrict__ x, const int* __restrict__ src,
                          const int* __restrict__ dst, bf16* __restrict__ out, int E) {
  int i = blockIdx.x * 256 + threadIdx.x;
  if (i >= E * 32) return;
  int e = i >> 5, c = i & 31;
  float v = 0.f;
  if (c < 3)      v = x[(size_t)src[e] * 3 + c];
  else if (c < 6) v = x[(size_t)dst[e] * 3 + (c - 3)];
  out[i] = (bf16)v;
}

__global__ void k_concat2(const float* __restrict__ nodes, const float* __restrict__ msg,
                          bf16* __restrict__ out, int Nn) {
  int i = blockIdx.x * 256 + threadIdx.x;
  if (i >= Nn * 256) return;
  int r = i >> 8, c = i & 255;
  float v = (c < 128) ? nodes[(size_t)r * 128 + c] : msg[(size_t)r * 128 + (c - 128)];
  out[i] = (bf16)v;
}

__global__ void k_concat3(const bf16* __restrict__ nodesb, const bf16* __restrict__ edgesb,
                          const int* __restrict__ src, const int* __restrict__ dst,
                          bf16* __restrict__ out, int E) {
  int i = blockIdx.x * 256 + threadIdx.x;
  if (i >= E * 384) return;
  int e = i / 384, c = i % 384;
  bf16 v;
  if (c < 128)      v = nodesb[(size_t)src[e] * 128 + c];
  else if (c < 256) v = nodesb[(size_t)dst[e] * 128 + (c - 128)];
  else              v = edgesb[(size_t)e * 128 + (c - 256)];
  out[i] = v;
}

__global__ void k_scatter_add(const float* __restrict__ edges, const int* __restrict__ dst,
                              float* __restrict__ msg, int E) {
  int i = blockIdx.x * 256 + threadIdx.x;
  if (i >= E * 128) return;
  int e = i >> 7, c = i & 127;
  atomicAdd(&msg[(size_t)dst[e] * 128 + c], edges[i]);
}

// ---- host orchestration ----------------------------------------------------

extern "C" void kernel_launch(void* const* d_in, const int* in_sizes, int n_in,
                              void* d_out, int out_size, void* d_ws, size_t ws_size,
                              hipStream_t stream) {
  (void)in_sizes; (void)n_in; (void)out_size; (void)ws_size;
  const int NN = 100000, NE = 400000, L = 128, H = 256, EMB = 12, ITERS = 4;

  const float* x     = (const float*)d_in[0];
  const int*   graph = (const int*)d_in[1];
  const int*   src   = graph;
  const int*   dst   = graph + NE;

  // Layer table in d_in flattening order (dict insertion order):
  // node_encoder(3), edge_encoder(3), output(3), then per cell: node(3), edge(3).
  int dims[33][2]; int li = 0;
  auto put = [&](int k, int n) { dims[li][0] = k; dims[li][1] = n; ++li; };
  put(3, H);      put(H, H); put(H, L);
  put(6, H);      put(H, H); put(H, L);
  put(L, H);      put(H, H); put(H, EMB);
  for (int c = 0; c < ITERS; ++c) {
    put(2 * L, H); put(H, H); put(H, L);   // node net
    put(3 * L, H); put(H, H); put(H, L);   // edge net
  }

  char* ws = (char*)d_ws;
  size_t cur = 0;
  auto carve = [&](size_t bytes) -> char* {
    cur = (cur + 255) & ~(size_t)255;
    char* p = ws + cur;
    cur += bytes;
    return p;
  };

  bf16* wt[33]; int Kp[33], Np[33];
  for (int j = 0; j < 33; ++j) {
    Kp[j] = (dims[j][0] + 31) & ~31;
    Np[j] = (dims[j][1] + 15) & ~15;
    wt[j] = (bf16*)carve((size_t)Kp[j] * Np[j] * sizeof(bf16));
  }
  bf16*  xin    = (bf16*)carve((size_t)NE * 384 * sizeof(bf16));
  bf16*  h1     = (bf16*)carve((size_t)NE * 256 * sizeof(bf16));
  bf16*  h2     = (bf16*)carve((size_t)NE * 256 * sizeof(bf16));
  float* nodes  = (float*)carve((size_t)NN * L * sizeof(float));
  bf16*  nodesb = (bf16*)carve((size_t)NN * L * sizeof(bf16));
  float* edges  = (float*)carve((size_t)NE * L * sizeof(float));
  bf16*  edgesb = (bf16*)carve((size_t)NE * L * sizeof(bf16));
  float* msg    = (float*)carve((size_t)NN * L * sizeof(float));

  // weight conversion (transpose + pad + fp32->bf16)
  for (int j = 0; j < 33; ++j) {
    int tot = Kp[j] * Np[j];
    k_cvt_wt<<<(tot + 255) / 256, 256, 0, stream>>>(
        (const float*)d_in[2 + j * 4], wt[j], dims[j][0], dims[j][1], Kp[j], Np[j]);
  }

  auto gemm = [&](int mode, const bf16* X, int j, const float* res,
                  float* of, bf16* ob, int M) {
    const float* b  = (const float*)d_in[2 + j * 4 + 1];
    const float* g  = (const float*)d_in[2 + j * 4 + 2];
    const float* be = (const float*)d_in[2 + j * 4 + 3];
    if (Np[j] == 256) {
      gemm_fused<8, 2, 0><<<(M + 63) / 64, 256, 0, stream>>>(
          X, wt[j], b, g, be, res, of, ob, M, Kp[j], EMB);
    } else if (Np[j] == 128) {
      if (mode == 1)
        gemm_fused<8, 1, 1><<<(M + 127) / 128, 256, 0, stream>>>(
            X, wt[j], b, g, be, res, of, ob, M, Kp[j], EMB);
      else
        gemm_fused<8, 1, 0><<<(M + 127) / 128, 256, 0, stream>>>(
            X, wt[j], b, g, be, res, of, ob, M, Kp[j], EMB);
    } else { // Np == 16, output head with L2 normalize
      gemm_fused<1, 1, 2><<<(M + 127) / 128, 256, 0, stream>>>(
          X, wt[j], b, g, be, nullptr, of, nullptr, M, Kp[j], EMB);
    }
  };

  // ---- node encoder
  k_node_in<<<(NN * 32 + 255) / 256, 256, 0, stream>>>(x, xin, NN);
  gemm(0, xin, 0, nullptr, nullptr, h1, NN);
  gemm(0, h1,  1, nullptr, nullptr, h2, NN);
  gemm(0, h2,  2, nullptr, nodes, nodesb, NN);

  // ---- edge encoder
  k_edge_in<<<(NE * 32 + 255) / 256, 256, 0, stream>>>(x, src, dst, xin, NE);
  gemm(0, xin, 3, nullptr, nullptr, h1, NE);
  gemm(0, h1,  4, nullptr, nullptr, h2, NE);
  gemm(0, h2,  5, nullptr, edges, edgesb, NE);

  // ---- message passing iterations
  for (int it = 0; it < ITERS; ++it) {
    int jb = 9 + it * 6;
    hipMemsetAsync(msg, 0, (size_t)NN * L * sizeof(float), stream);
    k_scatter_add<<<(NE * 128 + 255) / 256, 256, 0, stream>>>(edges, dst, msg, NE);

    k_concat2<<<(NN * 256 + 255) / 256, 256, 0, stream>>>(nodes, msg, xin, NN);
    gemm(0, xin, jb + 0, nullptr, nullptr, h1, NN);
    gemm(0, h1,  jb + 1, nullptr, nullptr, h2, NN);
    gemm(0, h2,  jb + 2, nodes, nodes, nodesb, NN);   // residual in place

    k_concat3<<<((size_t)NE * 384 + 255) / 256, 256, 0, stream>>>(
        nodesb, edgesb, src, dst, xin, NE);
    gemm(0, xin, jb + 3, nullptr, nullptr, h1, NE);
    gemm(0, h1,  jb + 4, nullptr, nullptr, h2, NE);
    gemm(1, h2,  jb + 5, edges, edges, edgesb, NE);   // tanh + residual
  }

  // ---- output MLP + normalize -> d_out[0 : NN*EMB]
  gemm(0, nodesb, 6, nullptr, nullptr, h1, NN);
  gemm(0, h1,     7, nullptr, nullptr, h2, NN);
  gemm(2, h2,     8, nullptr, (float*)d_out, nullptr, NN);

  // ---- nodes, edges into d_out tail
  float* out_nodes = (float*)d_out + (size_t)NN * EMB;
  float* out_edges = out_nodes + (size_t)NN * L;
  hipMemcpyAsync(out_nodes, nodes, (size_t)NN * L * sizeof(float),
                 hipMemcpyDeviceToDevice, stream);
  hipMemcpyAsync(out_edges, edges, (size_t)NE * L * sizeof(float),
                 hipMemcpyDeviceToDevice, stream);
}
